// SelfAttention_84095459655821
// MI455X (gfx1250) — matrix operand order repaired and checked by reading, not compiled, not run on hardware
//
#include <hip/hip_runtime.h>
#include <hip/hip_bf16.h>

#define DEVFN __device__ __forceinline__

typedef __attribute__((ext_vector_type(16))) __bf16 v16bf;
typedef __attribute__((ext_vector_type(8)))  __bf16 v8bf;
typedef __attribute__((ext_vector_type(8)))  float  v8f;
typedef __attribute__((ext_vector_type(4)))  unsigned int v4u;
typedef __attribute__((ext_vector_type(8)))  int v8i;
typedef __attribute__((ext_vector_type(4)))  int v4i;

constexpr int   BATCH = 4;
constexpr int   SEQ   = 4096;
constexpr int   DM    = 512;
constexpr int   DA    = 128;
constexpr int   DQKV  = 3 * DA;                 // 384
constexpr float SCALE = 0.08838834764831845f;   // 1/sqrt(128)
constexpr float LOG2E = 1.4426950408889634f;

#if __has_builtin(__builtin_amdgcn_tensor_load_to_lds) && \
    __has_builtin(__builtin_amdgcn_s_wait_tensorcnt)
#define USE_TDM 1
#else
#define USE_TDM 0
#endif

DEVFN __bf16 f2bf(float f) { return (__bf16)f; }

DEVFN v8f wmma_bf16(v16bf a, v16bf b, v8f c) {
  return __builtin_amdgcn_wmma_f32_16x16x32_bf16(false, a, false, b, (short)0, c,
                                                 false, false);
}

// A fragment (16x32 bf16): lane holds row=lane%16; elements 0..15 -> K=kb+16*half..
// (one contiguous 16-element run)
DEVFN v16bf a_frag_f32(const float* row, int kb, int half) {
  const float* p0 = row + kb + 16 * half;
  const float* p1 = row + kb + 16 * half + 8;
  v16bf a;
#pragma unroll
  for (int j = 0; j < 8; ++j) { a[j] = f2bf(p0[j]); a[8 + j] = f2bf(p1[j]); }
  return a;
}

DEVFN v16bf a_frag_bf16(const __bf16* row, int kb, int half) {
  v8bf lo = *(const v8bf*)(row + kb + 16 * half);
  v8bf hi = *(const v8bf*)(row + kb + 16 * half + 8);
  return __builtin_shufflevector(lo, hi, 0, 1, 2, 3, 4, 5, 6, 7,
                                 8, 9, 10, 11, 12, 13, 14, 15);
}

#if USE_TDM
// Issue a 2D TENSOR_LOAD_TO_LDS per CDNA5 D# spec (ISA 08_async_tensor §8.3-8.6).
// data_size=2 bytes, count=1, type=2(image), no gather/pad/iterate/multicast.
// This toolchain exposes the 6-arg builtin form:
//   (uint32x4 g0, int32x8 g1, int32x4 g2, int32x4 g3, int32x8, i32 cpol)
DEVFN void tdm_load_2d(unsigned lds_addr, const void* gptr,
                       unsigned td0, unsigned td1,
                       unsigned tile0, unsigned tile1,
                       unsigned long long stride0) {
  unsigned long long ga = (unsigned long long)(uintptr_t)gptr;
  // --- group 0 (128b): count | lds_addr | global_addr[56:0] | type=2 ---
  unsigned g0w0 = 1u;                                        // count=1
  unsigned g0w1 = lds_addr;
  unsigned g0w2 = (unsigned)ga;
  unsigned g0w3 = (unsigned)((ga >> 32) & 0x01FFFFFFull) | (2u << 30);
  // --- group 1 (256b) ---
  unsigned g1w0 = (1u << 16);                                // data_size=2B
  unsigned g1w1 = (td0 & 0xFFFFu) << 16;                     // tensor_dim0[15:0]
  unsigned g1w2 = (td0 >> 16) | ((td1 & 0xFFFFu) << 16);     // td0 hi | td1 lo
  unsigned g1w3 = (td1 >> 16) | ((tile0 & 0xFFFFu) << 16);   // td1 hi | tile_dim0
  unsigned g1w4 = tile1 & 0xFFFFu;                           // tile_dim1, tile_dim2=0
  unsigned g1w5 = (unsigned)(stride0 & 0xFFFFFFFFull);       // dim0_stride[31:0]
  unsigned g1w6 = (unsigned)((stride0 >> 32) & 0xFFFFull);   // dim0_stride[47:32]
  unsigned g1w7 = 0u;                                        // dim1_stride unused (2D)
  // Force descriptor words into SGPRs (values are wave-uniform).
  g0w0 = __builtin_amdgcn_readfirstlane(g0w0);
  g0w1 = __builtin_amdgcn_readfirstlane(g0w1);
  g0w2 = __builtin_amdgcn_readfirstlane(g0w2);
  g0w3 = __builtin_amdgcn_readfirstlane(g0w3);
  g1w0 = __builtin_amdgcn_readfirstlane(g1w0);
  g1w1 = __builtin_amdgcn_readfirstlane(g1w1);
  g1w2 = __builtin_amdgcn_readfirstlane(g1w2);
  g1w3 = __builtin_amdgcn_readfirstlane(g1w3);
  g1w4 = __builtin_amdgcn_readfirstlane(g1w4);
  g1w5 = __builtin_amdgcn_readfirstlane(g1w5);
  g1w6 = __builtin_amdgcn_readfirstlane(g1w6);
  g1w7 = __builtin_amdgcn_readfirstlane(g1w7);
  v4u G0 = {g0w0, g0w1, g0w2, g0w3};
  v8i G1 = {(int)g1w0, (int)g1w1, (int)g1w2, (int)g1w3,
            (int)g1w4, (int)g1w5, (int)g1w6, (int)g1w7};
  v4i Z4 = {0, 0, 0, 0};
  v8i Z8 = {0, 0, 0, 0, 0, 0, 0, 0};
  __builtin_amdgcn_tensor_load_to_lds(G0, G1, Z4, Z4, Z8, 0);
}
#endif

// ---------------------------------------------------------------------------
// Prep: W_qkv [512][384] f32 -> Wqt [384][512] bf16 ; W_out [128][512] -> Wot
// [512][128] bf16 (transposed so B fragments are contiguous per lane).
// ---------------------------------------------------------------------------
__global__ void prep_wqkv(const float* __restrict__ W, __bf16* __restrict__ Wqt) {
  int i = blockIdx.x * blockDim.x + threadIdx.x;
  if (i < DQKV * DM) {
    int n = i / DM, k = i % DM;
    Wqt[i] = f2bf(W[k * DQKV + n]);
  }
}

__global__ void prep_wout(const float* __restrict__ W, __bf16* __restrict__ Wot) {
  int i = blockIdx.x * blockDim.x + threadIdx.x;
  if (i < DM * DA) {
    int n = i / DA, k = i % DA;
    Wot[i] = f2bf(W[k * DM + n]);
  }
}

// ---------------------------------------------------------------------------
// QKV projection: [16384 x 512] @ [512 x 384] + bias. One wave per 16x64 tile
// (4 n-subtiles, K stepped by 32). Emits Q,K row-major bf16 and V transposed.
// ---------------------------------------------------------------------------
__global__ void __launch_bounds__(192)
qkv_gemm(const float* __restrict__ x, const __bf16* __restrict__ Wqt,
         const float* __restrict__ bqkv, __bf16* __restrict__ Q,
         __bf16* __restrict__ K, __bf16* __restrict__ VT) {
  const int wave = threadIdx.x >> 5;
  const int lane = threadIdx.x & 31;
  const int l16  = lane & 15;
  const int half = lane >> 4;
  const int m0   = blockIdx.x * 16;       // token tile
  const int n0   = wave * 64;             // output-column group

  const float* xrow = x + (size_t)(m0 + l16) * DM;
  v8f acc[4];
#pragma unroll
  for (int t = 0; t < 4; ++t) acc[t] = (v8f)0.0f;

  for (int kk = 0; kk < DM; kk += 32) {
    v16bf a = a_frag_f32(xrow, kk, half);
#pragma unroll
    for (int t = 0; t < 4; ++t) {
      const __bf16* wp = Wqt + (size_t)(n0 + t * 16 + l16) * DM + kk + 16 * half;
      v16bf b = *(const v16bf*)wp;
      acc[t] = wmma_bf16(a, b, acc[t]);
    }
  }

#pragma unroll
  for (int t = 0; t < 4; ++t) {
    const int c    = n0 + t * 16 + l16;
    const float bi = bqkv[c];
#pragma unroll
    for (int r = 0; r < 8; ++r) {
      const int row = m0 + r + 8 * half;  // global token 0..16383
      const float v = acc[t][r] + bi;
      if (c < DA) {
        Q[(size_t)row * DA + c] = f2bf(v);
      } else if (c < 2 * DA) {
        K[(size_t)row * DA + (c - DA)] = f2bf(v);
      } else {
        const int b_i = row / SEQ, pos = row % SEQ;
        VT[((size_t)b_i * DA + (c - 2 * DA)) * SEQ + pos] = f2bf(v);
      }
    }
  }
}

// ---------------------------------------------------------------------------
// Flash attention (causal, mask-before-scale as in reference).
// Block = 4 waves = 4 consecutive 16-row query tiles. K/V key blocks of 32 are
// staged into LDS once per block via the Tensor Data Mover (double-buffered,
// TENSORcnt-tracked), consumed 4x from LDS by the WMMA pipelines.
// ---------------------------------------------------------------------------
__global__ void __launch_bounds__(128)
flash_attn(const __bf16* __restrict__ Q, const __bf16* __restrict__ K,
           const __bf16* __restrict__ VT, const float* __restrict__ key_mask,
           __bf16* __restrict__ O) {
  __shared__ __attribute__((aligned(64))) __bf16 Kbuf[2][32][DA];   // 16 KB
  __shared__ __attribute__((aligned(64))) __bf16 Vbuf[2][DA][32];   // 16 KB
  __shared__ __attribute__((aligned(64))) __bf16 Plds[4][16 * 32];  // 4 KB

  const int wave = threadIdx.x >> 5;
  const int lane = threadIdx.x & 31;
  const int l16  = lane & 15;
  const int half = lane >> 4;
  const int bq   = blockIdx.y;                      // batch
  const int qt0  = blockIdx.x * 4;                  // first query tile of block
  const int qt   = qt0 + wave;
  const int m0   = qt * 16;

  const __bf16* Qb  = Q  + (size_t)bq * SEQ * DA;
  const __bf16* Kb  = K  + (size_t)bq * SEQ * DA;
  const __bf16* VTb = VT + (size_t)bq * DA * SEQ;
  __bf16*       Pw  = Plds[wave];

  // Q fragments for the whole tile (K=128 -> 4 chunks of 32)
  const __bf16* qrow = Qb + (size_t)(m0 + l16) * DA;
  v16bf qa[4];
#pragma unroll
  for (int ks = 0; ks < 4; ++ks) qa[ks] = a_frag_bf16(qrow, ks * 32, half);

  v8f acc[8];
#pragma unroll
  for (int t = 0; t < 8; ++t) acc[t] = (v8f)0.0f;
  float mrow[8], lrow[8];
#pragma unroll
  for (int r = 0; r < 8; ++r) { mrow[r] = -1e30f; lrow[r] = 0.0f; }

  const int jown = (m0 + 15) / 32 + 1;              // this wave's causal bound
  const int jmax = (qt0 * 16 + 63) / 32 + 1;        // block-wide bound (uniform)

#if USE_TDM
  if ((threadIdx.x >> 5) == 0) {                    // wave 0 drives the TDM
    tdm_load_2d((unsigned)(uintptr_t)&Kbuf[0][0][0], Kb, DA, 32, DA, 32, DA);
    tdm_load_2d((unsigned)(uintptr_t)&Vbuf[0][0][0], VTb, 32, DA, 32, DA, SEQ);
  }
#endif

  for (int jb = 0; jb < jmax; ++jb) {
    const int kb  = jb * 32;
    const int buf = jb & 1;

#if USE_TDM
    if ((threadIdx.x >> 5) == 0) {
      if (jb + 1 < jmax) {                          // prefetch next block
        tdm_load_2d((unsigned)(uintptr_t)&Kbuf[buf ^ 1][0][0],
                    Kb + (size_t)(kb + 32) * DA, DA, 32, DA, 32, DA);
        tdm_load_2d((unsigned)(uintptr_t)&Vbuf[buf ^ 1][0][0],
                    VTb + (kb + 32), 32, DA, 32, DA, SEQ);
        __builtin_amdgcn_s_wait_tensorcnt((short)2); // current pair done
      } else {
        __builtin_amdgcn_s_wait_tensorcnt((short)0);
      }
    }
#else
    {  // cooperative fallback copy: K tile 32x128, V tile 128x32 (bf16)
      const v8bf* gk = (const v8bf*)(Kb + (size_t)kb * DA);
      v8bf*       sk = (v8bf*)&Kbuf[buf][0][0];
      for (int idx = threadIdx.x; idx < (32 * DA) / 8; idx += blockDim.x)
        sk[idx] = gk[idx];
      for (int idx = threadIdx.x; idx < (32 * DA) / 8; idx += blockDim.x) {
        int dim = idx >> 2, ch = idx & 3;
        *(v8bf*)&Vbuf[buf][dim][ch * 8] =
            *(const v8bf*)(VTb + (size_t)dim * SEQ + kb + ch * 8);
      }
    }
#endif
    __syncthreads();                                // tile visible to all waves

    if (jb < jown) {                                // wave-uniform causal skip
      // ---- S = Q K^T (16x32), two 16-col subtiles, B from LDS ----
      v8f s[2];
      s[0] = (v8f)0.0f; s[1] = (v8f)0.0f;
#pragma unroll
      for (int t = 0; t < 2; ++t) {
        const __bf16* krow = &Kbuf[buf][t * 16 + l16][0];
#pragma unroll
        for (int ks = 0; ks < 4; ++ks) {
          v16bf bv = *(const v16bf*)(krow + ks * 32 + 16 * half);
          s[t] = wmma_bf16(qa[ks], bv, s[t]);
        }
      }

      // ---- mask (pre-scale), scale, online softmax ----
      float corr[8];
#pragma unroll
      for (int r = 0; r < 8; ++r) {
        const int rowg = m0 + r + 8 * half;
        const int c0   = kb + l16;
        const int c1   = kb + 16 + l16;
        float x0 = s[0][r] * SCALE; if (c0 > rowg) x0 = -1e30f;
        float x1 = s[1][r] * SCALE; if (c1 > rowg) x1 = -1e30f;

        float mx = fmaxf(x0, x1);
#pragma unroll
        for (int off = 1; off < 16; off <<= 1)
          mx = fmaxf(mx, __shfl_xor(mx, off, 32));
        const float mnew = fmaxf(mrow[r], mx);
        corr[r] = __builtin_exp2f((mrow[r] - mnew) * LOG2E);
        mrow[r] = mnew;

        const float p0 = __builtin_exp2f((x0 - mnew) * LOG2E);
        const float p1 = __builtin_exp2f((x1 - mnew) * LOG2E);
        float ps = p0 + p1;
#pragma unroll
        for (int off = 1; off < 16; off <<= 1)
          ps += __shfl_xor(ps, off, 32);
        lrow[r] = lrow[r] * corr[r] + ps;

        Pw[(r + 8 * half) * 32 + l16]      = f2bf(p0);
        Pw[(r + 8 * half) * 32 + 16 + l16] = f2bf(p1);
      }

#pragma unroll
      for (int t = 0; t < 8; ++t)
#pragma unroll
        for (int r = 0; r < 8; ++r) acc[t][r] *= corr[r];

      // ---- O += P V : A from LDS P tile, B from LDS V^T tile ----
      const __bf16* prow = Pw + l16 * 32;
      v8bf plo = *(const v8bf*)(prow + 16 * half);
      v8bf phi = *(const v8bf*)(prow + 16 * half + 8);
      v16bf pa = __builtin_shufflevector(plo, phi, 0, 1, 2, 3, 4, 5, 6, 7,
                                         8, 9, 10, 11, 12, 13, 14, 15);
#pragma unroll
      for (int t = 0; t < 8; ++t) {
        v16bf bv = *(const v16bf*)&Vbuf[buf][t * 16 + l16][16 * half];
        acc[t] = wmma_bf16(pa, bv, acc[t]);
      }
    }

    __syncthreads();                                // all consumed before reuse
  }

  // ---- normalize, apply key_mask, store O (bf16, row-major [B*N][128]) ----
  float km[8];
#pragma unroll
  for (int r = 0; r < 8; ++r) km[r] = key_mask[(size_t)bq * SEQ + m0 + r + 8 * half];
#pragma unroll
  for (int t = 0; t < 8; ++t) {
#pragma unroll
    for (int r = 0; r < 8; ++r) {
      const float o = (acc[t][r] / lrow[r]) * km[r];
      O[(size_t)(bq * SEQ + m0 + r + 8 * half) * DA + t * 16 + l16] = f2bf(o);
    }
  }
}

// ---------------------------------------------------------------------------
// Output projection: [16384 x 128] @ [128 x 512] + bias -> f32 out.
// ---------------------------------------------------------------------------
__global__ void __launch_bounds__(256)
out_gemm(const __bf16* __restrict__ O, const __bf16* __restrict__ Wot,
         const float* __restrict__ bout, float* __restrict__ out) {
  const int wave = threadIdx.x >> 5;
  const int lane = threadIdx.x & 31;
  const int l16  = lane & 15;
  const int half = lane >> 4;
  const int m0   = blockIdx.x * 16;
  const int n0   = wave * 64;

  const __bf16* orow = O + (size_t)(m0 + l16) * DA;
  v8f acc[4];
#pragma unroll
  for (int t = 0; t < 4; ++t) acc[t] = (v8f)0.0f;

#pragma unroll
  for (int kk = 0; kk < DA; kk += 32) {
    v16bf a = a_frag_bf16(orow, kk, half);
#pragma unroll
    for (int t = 0; t < 4; ++t) {
      const __bf16* wp = Wot + (size_t)(n0 + t * 16 + l16) * DA + kk + 16 * half;
      v16bf b = *(const v16bf*)wp;
      acc[t] = wmma_bf16(a, b, acc[t]);
    }
  }

#pragma unroll
  for (int t = 0; t < 4; ++t) {
    const int c    = n0 + t * 16 + l16;
    const float bi = bout[c];
#pragma unroll
    for (int r = 0; r < 8; ++r) {
      out[(size_t)(m0 + r + 8 * half) * DM + c] = acc[t][r] + bi;
    }
  }
}

// ---------------------------------------------------------------------------
extern "C" void kernel_launch(void* const* d_in, const int* in_sizes, int n_in,
                              void* d_out, int out_size, void* d_ws, size_t ws_size,
                              hipStream_t stream) {
  const float* x        = (const float*)d_in[0];
  const float* key_mask = (const float*)d_in[1];
  const float* W_qkv    = (const float*)d_in[2];
  const float* b_qkv    = (const float*)d_in[3];
  const float* W_out    = (const float*)d_in[4];
  const float* b_out    = (const float*)d_in[5];
  float* out            = (float*)d_out;

  char* p = (char*)d_ws;
  __bf16* Wqt = (__bf16*)p; p += (size_t)DQKV * DM * 2;       // 384*512
  __bf16* Wot = (__bf16*)p; p += (size_t)DM * DA * 2;         // 512*128
  __bf16* Q   = (__bf16*)p; p += (size_t)BATCH * SEQ * DA * 2;
  __bf16* K   = (__bf16*)p; p += (size_t)BATCH * SEQ * DA * 2;
  __bf16* VT  = (__bf16*)p; p += (size_t)BATCH * SEQ * DA * 2;
  __bf16* O   = (__bf16*)p; p += (size_t)BATCH * SEQ * DA * 2;

  prep_wqkv<<<(DQKV * DM + 255) / 256, 256, 0, stream>>>(W_qkv, Wqt);
  prep_wout<<<(DM * DA + 255) / 256, 256, 0, stream>>>(W_out, Wot);

  // QKV projection: 1024 row tiles x 6 waves (64 cols each)
  qkv_gemm<<<dim3(BATCH * SEQ / 16), dim3(192), 0, stream>>>(x, Wqt, b_qkv, Q, K, VT);

  // Flash attention: 4 query tiles per block (4 waves), per batch
  flash_attn<<<dim3(SEQ / 16 / 4, BATCH), dim3(128), 0, stream>>>(Q, K, VT, key_mask, O);

  // Output projection: 1024 row tiles x 8 waves (64 cols each)
  out_gemm<<<dim3(BATCH * SEQ / 16), dim3(256), 0, stream>>>(O, Wot, b_out, out);
}
